// MHCLiteBlockOld_81836306858485
// MI455X (gfx1250) — compile-verified, matrix operand order checked by reading, not run on hardware
//
#include <hip/hip_runtime.h>

// ---------------------------------------------------------------------------
// MHC-lite block for MI455X (gfx1250, wave32, WMMA).
// FFN (2 x 1024<->4096 GEMMs over 8192 tokens, 137 GFLOP) dominates -> WMMA
// bf16 16x16x32, async-to-LDS double-buffered mainloop (ASYNCcnt pipeline).
// ---------------------------------------------------------------------------

#define N_STREAMS 4
#define HIDDEN    1024
#define NC        4096
#define NFACT     24
#define TOKENS    8192           // B*S = 4*2048
#define DFF       4096
#define EPSF      1.1920929e-07f

typedef __attribute__((ext_vector_type(16))) __bf16 bf16x16;
typedef __attribute__((ext_vector_type(8)))  __bf16 bf16x8;
typedef __attribute__((ext_vector_type(4)))  __bf16 bf16x4;
typedef __attribute__((ext_vector_type(8)))  float  f32x8;

#define LDSS     40        // LDS row stride in bf16: 32 data + 8 pad = 80B (16B aligned, bank stride 20)
#define TILE_ELT (128 * LDSS)          // elements per LDS tile buffer
#define TILE_BYT (TILE_ELT * 2)        // 10240 bytes per buffer

// gfx1250 async copy: global -> LDS at the vDst byte address, 16B/lane, ASYNCcnt.
// LDS address = low 32 bits of the generic shared pointer (ISA: addr[31:0]).
__device__ __forceinline__ void async_ld_b128(unsigned int lds_byte, const void* g)
{
    asm volatile("global_load_async_to_lds_b128 %0, %1, off"
                 :: "v"(lds_byte), "v"(g) : "memory");
}

// ---------------------------------------------------------------------------
// Kernel 1: convert FFN weights f32 -> bf16 (row-major [N][K] kept as-is;
// B-fragment loads read W rows directly since Y = X @ W^T).
// ---------------------------------------------------------------------------
__global__ __launch_bounds__(256)
void mhc_cvt(const float* __restrict__ w1, const float* __restrict__ w2,
             __bf16* __restrict__ w1b, __bf16* __restrict__ w2b, int n)
{
    const int idx = (blockIdx.x * 256 + threadIdx.x) * 4;
    if (idx < n) {
        float4 a = *(const float4*)(w1 + idx);
        bf16x4 va; va[0] = (__bf16)a.x; va[1] = (__bf16)a.y;
        va[2] = (__bf16)a.z; va[3] = (__bf16)a.w;
        *(bf16x4*)(w1b + idx) = va;
        float4 b = *(const float4*)(w2 + idx);
        bf16x4 vb; vb[0] = (__bf16)b.x; vb[1] = (__bf16)b.y;
        vb[2] = (__bf16)b.z; vb[3] = (__bf16)b.w;
        *(bf16x4*)(w2b + idx) = vb;
    }
}

// ---------------------------------------------------------------------------
// Kernel 2: per-token prologue. One block = one token, 256 threads.
// rmsnorm -> 32-output projection GEMV -> gates/softmax/H_res ->
//   layer_input (bf16 + f32), h_post, and mixed = H_res @ x written to d_out.
// ---------------------------------------------------------------------------
__global__ __launch_bounds__(256)
void mhc_prologue(const float* __restrict__ x,
                  const float* __restrict__ alpha_pre,
                  const float* __restrict__ alpha_post,
                  const float* __restrict__ alpha_res,
                  const float* __restrict__ Wpre,  const float* __restrict__ bpre,
                  const float* __restrict__ Wpost, const float* __restrict__ bpost,
                  const float* __restrict__ Wres,  const float* __restrict__ bres,
                  const float* __restrict__ perm,
                  float*  __restrict__ out,        // [TOKENS][NC]  <- mixed
                  __bf16* __restrict__ Abf,        // [TOKENS][HIDDEN]
                  float*  __restrict__ Lf32,       // [TOKENS][HIDDEN]
                  float*  __restrict__ hpost_ws)   // [TOKENS][4]
{
    const int tok = blockIdx.x;
    const int t   = threadIdx.x;
    const float* xt = x + (size_t)tok * NC;

    __shared__ float red[256];
    __shared__ float part[32 * 257];   // 257-stride -> conflict-free column sums
    __shared__ float sH[16];
    __shared__ float sHpre[4];
    __shared__ float sHpost[4];
    __shared__ float sScale;

    // each thread owns 16 contiguous x values
    float xv[16];
    {
        const float4* x4 = (const float4*)(xt + t * 16);
#pragma unroll
        for (int i = 0; i < 4; ++i) {
            float4 v = x4[i];
            xv[i * 4 + 0] = v.x; xv[i * 4 + 1] = v.y;
            xv[i * 4 + 2] = v.z; xv[i * 4 + 3] = v.w;
        }
    }
    float ss = 0.f;
#pragma unroll
    for (int i = 0; i < 16; ++i) ss += xv[i] * xv[i];
    red[t] = ss;
    __syncthreads();
    for (int s = 128; s > 0; s >>= 1) {
        if (t < s) red[t] += red[t + s];
        __syncthreads();
    }
    if (t == 0) sScale = rsqrtf(red[0] * (1.0f / NC) + EPSF);
    __syncthreads();
    const float scale = sScale;

    // 32 projection partials (pre:0-3, post:4-7, res:8-31); linear in scale
    {
        const int base = t * 16;
        for (int o = 0; o < 32; ++o) {
            const float* w = (o < 4) ? (Wpre + o * NC)
                           : (o < 8) ? (Wpost + (o - 4) * NC)
                                     : (Wres + (o - 8) * NC);
            const float4* w4 = (const float4*)(w + base);
            float p = 0.f;
#pragma unroll
            for (int i = 0; i < 4; ++i) {
                float4 wv = w4[i];
                p += xv[i * 4 + 0] * wv.x + xv[i * 4 + 1] * wv.y +
                     xv[i * 4 + 2] * wv.z + xv[i * 4 + 3] * wv.w;
            }
            part[o * 257 + t] = p * scale;
        }
    }
    __syncthreads();
    if (t < 32) {
        float s = 0.f;
        for (int i = 0; i < 256; ++i) s += part[t * 257 + i];
        part[t * 257] = s;
    }
    __syncthreads();
    if (t == 0) {
        const float ap = alpha_pre[0], apo = alpha_post[0], ar = alpha_res[0];
#pragma unroll
        for (int n = 0; n < 4; ++n) {
            sHpre[n]  = 1.0f / (1.0f + expf(-ap  * (part[n * 257] + bpre[n])));
            sHpost[n] = 2.0f / (1.0f + expf(-apo * (part[(4 + n) * 257] + bpost[n])));
        }
        float r[NFACT];
        float mx = -1e30f;
        for (int p = 0; p < NFACT; ++p) {
            r[p] = ar * (part[(8 + p) * 257] + bres[p]);
            mx = fmaxf(mx, r[p]);
        }
        float sum = 0.f;
        for (int p = 0; p < NFACT; ++p) { r[p] = expf(r[p] - mx); sum += r[p]; }
        const float inv = 1.0f / sum;
        for (int e = 0; e < 16; ++e) {
            float h = 0.f;
            for (int p = 0; p < NFACT; ++p) h += r[p] * perm[p * 16 + e];
            sH[e] = h * inv;
        }
    }
    __syncthreads();

    // layer_input[c] = sum_n h_pre[n] * x[n][c]; thread t owns c = 4t..4t+3
    {
        const float4* xs4 = (const float4*)xt;
        float lx = 0.f, ly = 0.f, lz = 0.f, lw = 0.f;
#pragma unroll
        for (int n = 0; n < N_STREAMS; ++n) {
            float4 v = xs4[n * (HIDDEN / 4) + t];
            const float h = sHpre[n];
            lx += h * v.x; ly += h * v.y; lz += h * v.z; lw += h * v.w;
        }
        float4 li; li.x = lx; li.y = ly; li.z = lz; li.w = lw;
        ((float4*)(Lf32 + (size_t)tok * HIDDEN))[t] = li;
        bf16x4 lb; lb[0] = (__bf16)lx; lb[1] = (__bf16)ly;
        lb[2] = (__bf16)lz; lb[3] = (__bf16)lw;
        *(bf16x4*)(Abf + (size_t)tok * HIDDEN + t * 4) = lb;
    }
    if (t < 4) hpost_ws[(size_t)tok * 4 + t] = sHpost[t];

    // mixed[i][c] = sum_j H_res[i][j]*x[j][c]; thread owns stream i=t>>6, 16 channels
    {
        const int i  = t >> 6;
        const int c0 = (t & 63) * 16;
        const float4* xs4 = (const float4*)xt;
        float acc[16];
#pragma unroll
        for (int k = 0; k < 16; ++k) acc[k] = 0.f;
#pragma unroll
        for (int j = 0; j < N_STREAMS; ++j) {
            const float h = sH[i * 4 + j];
#pragma unroll
            for (int v = 0; v < 4; ++v) {
                float4 xw = xs4[(j * HIDDEN + c0) / 4 + v];
                acc[v * 4 + 0] += h * xw.x; acc[v * 4 + 1] += h * xw.y;
                acc[v * 4 + 2] += h * xw.z; acc[v * 4 + 3] += h * xw.w;
            }
        }
        float4* o4 = (float4*)(out + (size_t)tok * NC + i * HIDDEN + c0);
#pragma unroll
        for (int v = 0; v < 4; ++v) {
            float4 ov; ov.x = acc[v * 4 + 0]; ov.y = acc[v * 4 + 1];
            ov.z = acc[v * 4 + 2]; ov.w = acc[v * 4 + 3];
            o4[v] = ov;
        }
    }
}

// ---------------------------------------------------------------------------
// Shared WMMA mainloop: C[128x128] block tile, 8 waves, each wave 64x32
// (4 M-tiles x 2 N-tiles of 16x16), K stepped by 32.
// Double-buffered LDS fed by GLOBAL_LOAD_ASYNC_TO_LDS_B128 (ASYNCcnt):
//   preload tile0; each iter issues tile i+1 into the other buffer, waits
//   asynccnt<=4 (tile i landed, tile i+1 in flight behind the WMMAs).
// A: [M][K] bf16 row-major (lane = M row, elems = K per ISA 16-bit A layout)
// B: [N][K] bf16 row-major (Y = X @ W^T -> B fragment = W row, lane = N)
// ---------------------------------------------------------------------------
__device__ __forceinline__ void gemm_mainloop(
    const __bf16* __restrict__ A, const __bf16* __restrict__ B,
    int lda, int ldb, int K, int mBlock, int nBlock,
    __bf16* ldsA, __bf16* ldsB, f32x8 acc[4][2])
{
    const int t     = threadIdx.x;
    const int lane  = t & 31;
    const int wave  = t >> 5;
    const int waveM = wave >> 2;        // 0..1
    const int waveN = wave & 3;         // 0..3
    const int lr    = t >> 1;           // 0..127 cooperative-copy row
    const int lc    = (t & 1) * 16;     // 0 / 16  cooperative-copy col
    const int kHalf = (lane >> 4) * 8;  // ISA: lanes 16-31 hold K+8
    const int row16 = lane & 15;

    const __bf16* gA = A + (size_t)(mBlock + lr) * lda + lc;
    const __bf16* gB = B + (size_t)(nBlock + lr) * ldb + lc;

    // per-thread LDS byte destinations (low 32 bits of generic shared ptr)
    const unsigned int lA0 = (unsigned int)(size_t)(ldsA + lr * LDSS + lc);
    const unsigned int lB0 = (unsigned int)(size_t)(ldsB + lr * LDSS + lc);

    const int nTiles = K >> 5;

    // preload tile 0 into buffer 0
    async_ld_b128(lA0,      gA);
    async_ld_b128(lA0 + 16, gA + 8);
    async_ld_b128(lB0,      gB);
    async_ld_b128(lB0 + 16, gB + 8);

    for (int i = 0; i < nTiles; ++i) {
        const int cur = i & 1;
        if (i + 1 < nTiles) {
            const __bf16* nA = gA + (i + 1) * 32;
            const __bf16* nB = gB + (i + 1) * 32;
            const unsigned int nxt = (cur ^ 1) * TILE_BYT;
            async_ld_b128(lA0 + nxt,      nA);
            async_ld_b128(lA0 + nxt + 16, nA + 8);
            async_ld_b128(lB0 + nxt,      nB);
            async_ld_b128(lB0 + nxt + 16, nB + 8);
            asm volatile("s_wait_asynccnt 0x4" ::: "memory");
        } else {
            asm volatile("s_wait_asynccnt 0x0" ::: "memory");
        }
        __syncthreads();   // tile i visible to all waves

        const __bf16* bufA = ldsA + cur * TILE_ELT;
        const __bf16* bufB = ldsB + cur * TILE_ELT;

        bf16x16 af[4];
#pragma unroll
        for (int mi = 0; mi < 4; ++mi) {
            const __bf16* p = bufA + (waveM * 64 + mi * 16 + row16) * LDSS + kHalf;
            bf16x8 lo = *(const bf16x8*)p;        // K = kHalf .. kHalf+7
            bf16x8 hi = *(const bf16x8*)(p + 16); // K = kHalf+16 .. kHalf+23
            af[mi] = __builtin_shufflevector(lo, hi,
                     0, 1, 2, 3, 4, 5, 6, 7, 8, 9, 10, 11, 12, 13, 14, 15);
        }
        bf16x16 bfr[2];
#pragma unroll
        for (int ni = 0; ni < 2; ++ni) {
            const __bf16* p = bufB + (waveN * 32 + ni * 16 + row16) * LDSS + kHalf;
            bf16x8 lo = *(const bf16x8*)p;
            bf16x8 hi = *(const bf16x8*)(p + 16);
            bfr[ni] = __builtin_shufflevector(lo, hi,
                      0, 1, 2, 3, 4, 5, 6, 7, 8, 9, 10, 11, 12, 13, 14, 15);
        }
#pragma unroll
        for (int mi = 0; mi < 4; ++mi)
#pragma unroll
            for (int ni = 0; ni < 2; ++ni)
                acc[mi][ni] = __builtin_amdgcn_wmma_f32_16x16x32_bf16(
                    false, af[mi], false, bfr[ni], (short)0, acc[mi][ni],
                    false, false);
        __syncthreads();   // all waves done reading buf[cur] before it is refilled
    }
}

// ---------------------------------------------------------------------------
// Kernel 3: H = gelu_tanh(layer_input @ W1^T + b1)  [8192,1024]x[4096,1024]
// ---------------------------------------------------------------------------
__global__ __launch_bounds__(256)
void mhc_ffn1(const __bf16* __restrict__ A,   // [TOKENS][HIDDEN]
              const __bf16* __restrict__ W1,  // [DFF][HIDDEN]
              const float*  __restrict__ b1,  // [DFF]
              __bf16* __restrict__ H)         // [TOKENS][DFF]
{
    __shared__ __bf16 ldsA[2 * TILE_ELT];
    __shared__ __bf16 ldsB[2 * TILE_ELT];
    const int mBlock = blockIdx.y * 128;
    const int nBlock = blockIdx.x * 128;

    f32x8 acc[4][2];
    const f32x8 zero = {0.f, 0.f, 0.f, 0.f, 0.f, 0.f, 0.f, 0.f};
#pragma unroll
    for (int mi = 0; mi < 4; ++mi)
#pragma unroll
        for (int ni = 0; ni < 2; ++ni) acc[mi][ni] = zero;

    gemm_mainloop(A, W1, HIDDEN, HIDDEN, HIDDEN, mBlock, nBlock, ldsA, ldsB, acc);

    const int lane  = threadIdx.x & 31;
    const int wave  = threadIdx.x >> 5;
    const int mBase = mBlock + (wave >> 2) * 64 + 8 * (lane >> 4);
    const int nBase = nBlock + (wave & 3) * 32 + (lane & 15);
#pragma unroll
    for (int mi = 0; mi < 4; ++mi) {
#pragma unroll
        for (int ni = 0; ni < 2; ++ni) {
            const int n = nBase + ni * 16;
            const float bias = b1[n];
#pragma unroll
            for (int j = 0; j < 8; ++j) {
                const int m = mBase + mi * 16 + j;
                float y = acc[mi][ni][j] + bias;
                float u = 0.7978845608028654f * (y + 0.044715f * y * y * y);
                float g = 0.5f * y * (1.0f + tanhf(u));
                H[(size_t)m * DFF + n] = (__bf16)g;
            }
        }
    }
}

// ---------------------------------------------------------------------------
// Kernel 4: Y = H @ W2^T + b2; delta = Y - layer_input;
//           out[m][s][:] += h_post[m][s] * delta   (out pre-filled with mixed)
// ---------------------------------------------------------------------------
__global__ __launch_bounds__(256)
void mhc_ffn2(const __bf16* __restrict__ H,     // [TOKENS][DFF]
              const __bf16* __restrict__ W2,    // [HIDDEN][DFF]
              const float*  __restrict__ b2,    // [HIDDEN]
              const float*  __restrict__ Lf32,  // [TOKENS][HIDDEN]
              const float*  __restrict__ hpost, // [TOKENS][4]
              float* __restrict__ out)          // [TOKENS][NC]
{
    __shared__ __bf16 ldsA[2 * TILE_ELT];
    __shared__ __bf16 ldsB[2 * TILE_ELT];
    const int mBlock = blockIdx.y * 128;
    const int nBlock = blockIdx.x * 128;

    f32x8 acc[4][2];
    const f32x8 zero = {0.f, 0.f, 0.f, 0.f, 0.f, 0.f, 0.f, 0.f};
#pragma unroll
    for (int mi = 0; mi < 4; ++mi)
#pragma unroll
        for (int ni = 0; ni < 2; ++ni) acc[mi][ni] = zero;

    gemm_mainloop(H, W2, DFF, DFF, DFF, mBlock, nBlock, ldsA, ldsB, acc);

    const int lane  = threadIdx.x & 31;
    const int wave  = threadIdx.x >> 5;
    const int mBase = mBlock + (wave >> 2) * 64 + 8 * (lane >> 4);
    const int nBase = nBlock + (wave & 3) * 32 + (lane & 15);
#pragma unroll
    for (int mi = 0; mi < 4; ++mi) {
#pragma unroll
        for (int ni = 0; ni < 2; ++ni) {
            const int n = nBase + ni * 16;
            const float bias = b2[n];
#pragma unroll
            for (int j = 0; j < 8; ++j) {
                const int m = mBase + mi * 16 + j;
                const float delta =
                    acc[mi][ni][j] + bias - Lf32[(size_t)m * HIDDEN + n];
                const float4 hp = *(const float4*)(hpost + (size_t)m * 4);
                float* o = out + (size_t)m * NC + n;
                o[0 * HIDDEN] += hp.x * delta;
                o[1 * HIDDEN] += hp.y * delta;
                o[2 * HIDDEN] += hp.z * delta;
                o[3 * HIDDEN] += hp.w * delta;
            }
        }
    }
}

// ---------------------------------------------------------------------------
// Launch. Workspace layout (bytes, all offsets 256B aligned):
//   Abf   @ 0          16,777,216   layer_input bf16 [8192][1024]
//   Lf32  @ 16777216   33,554,432   layer_input f32
//   hpost @ 50331648      131,072
//   W1b   @ 50462720    8,388,608
//   W2b   @ 58851328    8,388,608
//   Hb    @ 67239936   67,108,864   hidden acts bf16 [8192][4096]
//   total ~134.3 MB
// ---------------------------------------------------------------------------
extern "C" void kernel_launch(void* const* d_in, const int* in_sizes, int n_in,
                              void* d_out, int out_size, void* d_ws, size_t ws_size,
                              hipStream_t stream)
{
    const float* x        = (const float*)d_in[0];
    const float* a_pre    = (const float*)d_in[1];
    const float* a_post   = (const float*)d_in[2];
    const float* a_res    = (const float*)d_in[3];
    const float* Wpre     = (const float*)d_in[4];
    const float* bpre     = (const float*)d_in[5];
    const float* Wpost    = (const float*)d_in[6];
    const float* bpost    = (const float*)d_in[7];
    const float* Wres     = (const float*)d_in[8];
    const float* bres     = (const float*)d_in[9];
    const float* ffn_w1   = (const float*)d_in[10];
    const float* ffn_b1   = (const float*)d_in[11];
    const float* ffn_w2   = (const float*)d_in[12];
    const float* ffn_b2   = (const float*)d_in[13];
    const float* perm     = (const float*)d_in[14];

    char* ws = (char*)d_ws;
    __bf16* Abf   = (__bf16*)(ws + 0);
    float*  Lf32  = (float*) (ws + 16777216);
    float*  hpost = (float*) (ws + 50331648);
    __bf16* W1b   = (__bf16*)(ws + 50462720);
    __bf16* W2b   = (__bf16*)(ws + 58851328);
    __bf16* Hb    = (__bf16*)(ws + 67239936);
    float*  out   = (float*)d_out;

    mhc_cvt<<<4096, 256, 0, stream>>>(ffn_w1, ffn_w2, W1b, W2b, DFF * HIDDEN);

    mhc_prologue<<<TOKENS, 256, 0, stream>>>(
        x, a_pre, a_post, a_res, Wpre, bpre, Wpost, bpost, Wres, bres, perm,
        out, Abf, Lf32, hpost);

    dim3 g1(DFF / 128, TOKENS / 128);      // (32, 64)
    mhc_ffn1<<<g1, 256, 0, stream>>>(Abf, W1b, ffn_b1, Hb);

    dim3 g2(HIDDEN / 128, TOKENS / 128);   // (8, 64)
    mhc_ffn2<<<g2, 256, 0, stream>>>(Hb, W2b, ffn_b2, Lf32, hpost, out);
}
